// ConvStackedTemporalGCN_67242007986542
// MI455X (gfx1250) — matrix operand order repaired and checked by reading, not compiled, not run on hardware
//
#include <hip/hip_runtime.h>
#include <hip/hip_bf16.h>

// CDNA5 (gfx1250) wave32 WMMA types
typedef __attribute__((ext_vector_type(16))) __bf16 v16bf;
typedef __attribute__((ext_vector_type(8)))  __bf16 v8bf;
typedef __attribute__((ext_vector_type(8)))  float  v8f;

static constexpr int N_   = 10000;
static constexpr int F_   = 128;
static constexpr int P_   = 4;
static constexpr int E_   = 160000;
static constexpr int C_   = 512;
static constexpr int HID_ = 256;

static constexpr int BM = 128, BN = 64, BK = 32;
static constexpr int LDP = BK + 8;   // LDS row stride (keeps 16B alignment, 80B rows)
static constexpr int ACT_NONE = 0, ACT_RELU = 1, ACT_SIG = 2, ACT_TANH = 3;

__device__ __forceinline__ float bfbits2f(unsigned s) {
    union { unsigned u; float f; } c; c.u = s << 16; return c.f;
}

// Async DMA of 16 bytes global -> LDS (gfx1250 GLOBAL_LOAD_ASYNC_TO_LDS_B128,
// tracked by ASYNCcnt). ldsOff is the wave-relative LDS byte address = low 32
// bits of the flat pointer (ISA 10.2 aperture truncation).
__device__ __forceinline__ void async_g2l_b128(unsigned ldsOff, const void* gptr) {
    asm volatile("global_load_async_to_lds_b128 %0, %1, off"
                 :: "v"(ldsOff), "v"((unsigned long long)(size_t)gptr)
                 : "memory");
}
__device__ __forceinline__ void wait_asynccnt0() {
    asm volatile("s_wait_asynccnt 0x0" ::: "memory");
}

// ---------------------------------------------------------------------------
// bf16 GEMM: C = act(A[MxK] @ W[KxN] + bias) with W supplied pre-TRANSPOSED
// as Bt[N x K] (ldbt = K). fp32 accumulate via v_wmma_f32_16x16x32_bf16.
// Block = 256 threads = 8 wave32 waves arranged 4(M) x 2(N); each wave owns a
// 32x32 tile (2x2 WMMA tiles). Tiles stage into LDS via the CDNA5 async
// global->LDS DMA path; fragments are two aligned ds_load_b128 per lane.
// ---------------------------------------------------------------------------
__global__ __launch_bounds__(256)
void gemm_bf16_wmma_kernel(const __bf16* __restrict__ A, int lda,
                           const __bf16* __restrict__ Bt, int ldbt,
                           const float* __restrict__ bias,
                           float* __restrict__ Cf, int ldcf,
                           __bf16* __restrict__ Cb, int ldcb,
                           int M, int Ncols, int K, int act)
{
    __shared__ __bf16 As[BM][LDP];   // [m][k]
    __shared__ __bf16 Bs[BN][LDP];   // [n][k]  (weights pre-transposed)

    const int tid  = threadIdx.x;
    const int lane = tid & 31;
    const int wave = tid >> 5;
    const int wm   = wave & 3;   // 4 wave-rows of M (32 each)
    const int wn   = wave >> 2;  // 2 wave-cols of N (32 each)
    const int bm0  = blockIdx.x * BM;
    const int bn0  = blockIdx.y * BN;

    v8f acc[2][2];
    #pragma unroll
    for (int i = 0; i < 2; ++i)
        #pragma unroll
        for (int j = 0; j < 2; ++j)
            #pragma unroll
            for (int r = 0; r < 8; ++r) acc[i][j][r] = 0.0f;

    const int ml   = lane & 15;
    const int half = lane >> 4;

    // Staging coordinates (row-granular bounds, 16B DMA moves)
    const int ar  = tid >> 1;            // A row 0..127
    const int ac0 = (tid & 1) * 16;      // A col 0 or 16
    const int agr = bm0 + ar;
    const bool aIn = (agr < M);
    const int bn  = tid >> 2;            // B row (n) 0..63
    const int bk0 = (tid & 3) * 8;       // B col (k) 0,8,16,24

    const unsigned ldsA = (unsigned)(size_t)&As[ar][ac0];
    const unsigned ldsB = (unsigned)(size_t)&Bs[bn][bk0];
    const __bf16* gA = A  + (long)agr * lda + ac0;
    const __bf16* gB = Bt + (long)(bn0 + bn) * ldbt + bk0;

    for (int k0 = 0; k0 < K; k0 += BK) {
        // ---- stage A (BM x BK): 32B per thread via async DMA ----
        if (aIn) {
            async_g2l_b128(ldsA,      gA + k0);
            async_g2l_b128(ldsA + 16, gA + k0 + 8);
        } else {
            uint4 z = {0u, 0u, 0u, 0u};
            *(uint4*)&As[ar][ac0]     = z;   // disjoint rows from async writes
            *(uint4*)&As[ar][ac0 + 8] = z;
        }
        // ---- stage Bt (BN x BK): 16B per thread via async DMA ----
        async_g2l_b128(ldsB, gB + k0);
        if (k0 + BK < K) {
            if (aIn) __builtin_prefetch(gA + k0 + BK, 0, 3);
            __builtin_prefetch(gB + k0 + BK, 0, 3);
        }
        wait_asynccnt0();
        __syncthreads();

        // ---- fragments (ISA 7.12.2): two aligned b128 LDS loads each ----
        v16bf afr[2], bfr[2];
        #pragma unroll
        for (int mt = 0; mt < 2; ++mt) {
            const int row = wm * 32 + mt * 16 + ml;
            // element i (0..7):  k = half*8 + i      -> contiguous
            // element i (8..15): k = 16 + half*8 + (i-8) -> contiguous
            v8bf lo = *(const v8bf*)&As[row][half * 8];
            v8bf hi = *(const v8bf*)&As[row][16 + half * 8];
            afr[mt] = __builtin_shufflevector(lo, hi, 0, 1, 2, 3, 4, 5, 6, 7,
                                              8, 9, 10, 11, 12, 13, 14, 15);
        }
        #pragma unroll
        for (int nt = 0; nt < 2; ++nt) {
            const int col = wn * 32 + nt * 16 + ml;
            // element i (0..7): k = half*8 + i; element i (8..15): k = 16 + half*8 + (i-8)
            v8bf lo = *(const v8bf*)&Bs[col][half * 8];
            v8bf hi = *(const v8bf*)&Bs[col][16 + half * 8];
            bfr[nt] = __builtin_shufflevector(lo, hi, 0, 1, 2, 3, 4, 5, 6, 7,
                                              8, 9, 10, 11, 12, 13, 14, 15);
        }
        #pragma unroll
        for (int mt = 0; mt < 2; ++mt)
            #pragma unroll
            for (int nt = 0; nt < 2; ++nt)
                acc[mt][nt] = __builtin_amdgcn_wmma_f32_16x16x32_bf16(
                    false, afr[mt], false, bfr[nt], (short)0, acc[mt][nt],
                    false, false);
        __syncthreads();
    }

    // Epilogue: C/D layout (lane n = lane&15; VGPR r -> m = (lane>>4)*8 + r)
    #pragma unroll
    for (int mt = 0; mt < 2; ++mt) {
        #pragma unroll
        for (int nt = 0; nt < 2; ++nt) {
            const int gn = bn0 + wn * 32 + nt * 16 + ml;
            const float bv = bias ? bias[gn] : 0.0f;
            #pragma unroll
            for (int r = 0; r < 8; ++r) {
                const int gm = bm0 + wm * 32 + mt * 16 + half * 8 + r;
                if (gm < M) {
                    float v = acc[mt][nt][r] + bv;
                    if (act == ACT_RELU)      v = fmaxf(v, 0.0f);
                    else if (act == ACT_SIG)  v = 1.0f / (1.0f + __expf(-v));
                    else if (act == ACT_TANH) v = tanhf(v);
                    if (Cf) Cf[(long)gm * ldcf + gn] = v;
                    if (Cb) Cb[(long)gm * ldcb + gn] = (__bf16)v;
                }
            }
        }
    }
}

// ---------------------------------------------------------------------------
// Elementwise / graph kernels
// ---------------------------------------------------------------------------
__global__ void zero_f32_kernel(float* __restrict__ p, long n) {
    long i = blockIdx.x * (long)blockDim.x + threadIdx.x;
    if (i < n) p[i] = 0.0f;
}

// transposed fp32 -> bf16 weight convert: out[n*K+k] = (bf16)in[k*N+n]
__global__ void f32_to_bf16_t_kernel(const float* __restrict__ in,
                                     __bf16* __restrict__ out, int K, int Nc) {
    long gid = blockIdx.x * (long)blockDim.x + threadIdx.x;
    if (gid >= (long)K * Nc) return;
    int n = (int)(gid / K), k = (int)(gid % K);
    out[gid] = (__bf16)in[(long)k * Nc + n];
}

__global__ void slice_xp_kernel(const float* __restrict__ x,
                                __bf16* __restrict__ Xp, int p) {
    long i = blockIdx.x * (long)blockDim.x + threadIdx.x;
    if (i < (long)N_ * F_) Xp[i] = (__bf16)x[i * P_ + p];
}

__global__ void deg_scatter_kernel(const float* __restrict__ ew,
                                   const int* __restrict__ dst,
                                   float* __restrict__ deg) {
    int e = blockIdx.x * blockDim.x + threadIdx.x;
    if (e < E_) atomicAdd(&deg[dst[e]], ew[e]);
}

__global__ void dinv_kernel(const float* __restrict__ deg,
                            float* __restrict__ dinv,
                            float* __restrict__ selfn) {
    int i = blockIdx.x * blockDim.x + threadIdx.x;
    if (i < N_) {
        float d = deg[i] + 1.0f;           // + self-loop weight
        float dv = rsqrtf(d);
        dinv[i]  = dv;
        selfn[i] = dv * dv;                // norm of self-loop edge
    }
}

__global__ void edge_norm_kernel(const float* __restrict__ ew,
                                 const int* __restrict__ src,
                                 const int* __restrict__ dst,
                                 const float* __restrict__ dinv,
                                 float* __restrict__ normv) {
    int e = blockIdx.x * blockDim.x + threadIdx.x;
    if (e < E_) normv[e] = dinv[src[e]] * ew[e] * dinv[dst[e]];
}

// acc[dst] += t[src] * norm  (4 channels/thread; one b64 bf16 gather, fp32 atomics)
__global__ void scatter_kernel(const __bf16* __restrict__ t,
                               const int* __restrict__ src,
                               const int* __restrict__ dst,
                               const float* __restrict__ normv,
                               float* __restrict__ acc) {
    long gid = blockIdx.x * (long)blockDim.x + threadIdx.x;
    constexpr int PER = C_ / 4;
    if (gid >= (long)E_ * PER) return;
    int e = (int)(gid / PER);
    int c = (int)(gid % PER) * 4;
    int s = src[e], d = dst[e];
    float w = normv[e];
    uint2 pv = *(const uint2*)(t + (long)s * C_ + c);
    float* ap = acc + (long)d * C_ + c;
    atomicAdd(ap + 0, bfbits2f(pv.x & 0xffffu) * w);
    atomicAdd(ap + 1, bfbits2f(pv.x >> 16) * w);
    atomicAdd(ap + 2, bfbits2f(pv.y & 0xffffu) * w);
    atomicAdd(ap + 3, bfbits2f(pv.y >> 16) * w);
}

// g = acc + t[i]*selfnorm[i] + bias ; write bf16 (strided) and optional fp32
__global__ void gcn_finish_kernel(const float* __restrict__ acc,
                                  const __bf16* __restrict__ t,
                                  const float* __restrict__ selfn,
                                  const float* __restrict__ bias,
                                  __bf16* __restrict__ outB, int ldOutB,
                                  float* __restrict__ outF) {
    long gid = blockIdx.x * (long)blockDim.x + threadIdx.x;
    if (gid >= (long)N_ * C_) return;
    int i = (int)(gid / C_), c = (int)(gid % C_);
    float v = acc[gid] + (float)t[gid] * selfn[i] + bias[c];
    outB[(long)i * ldOutB + c] = (__bf16)v;
    if (outF) outF[gid] = v;
}

// CC[:, C:2C] = bf16(H) or bf16(H*R)
__global__ void concat_right_kernel(const float* __restrict__ H,
                                    const float* __restrict__ R,
                                    __bf16* __restrict__ CC) {
    long gid = blockIdx.x * (long)blockDim.x + threadIdx.x;
    if (gid >= (long)N_ * C_) return;
    int i = (int)(gid / C_), c = (int)(gid % C_);
    float v = H[gid];
    if (R) v *= R[gid];
    CC[(long)i * (2 * C_) + C_ + c] = (__bf16)v;
}

// Haccum += probs[p] * (Z*H + (1-Z)*Ht)
__global__ void gate_kernel(const float* __restrict__ Z,
                            const float* __restrict__ H,
                            const float* __restrict__ Ht,
                            const float* __restrict__ probs, int p,
                            float* __restrict__ Hacc) {
    long gid = blockIdx.x * (long)blockDim.x + threadIdx.x;
    if (gid >= (long)N_ * C_) return;
    float z = Z[gid];
    Hacc[gid] += probs[p] * (z * H[gid] + (1.0f - z) * Ht[gid]);
}

__global__ void relu_to_bf16_kernel(const float* __restrict__ H,
                                    __bf16* __restrict__ out) {
    long gid = blockIdx.x * (long)blockDim.x + threadIdx.x;
    if (gid < (long)N_ * C_) out[gid] = (__bf16)fmaxf(H[gid], 0.0f);
}

__global__ void softmax_attn_kernel(const float* __restrict__ attn,
                                    float* __restrict__ probs) {
    if (blockIdx.x == 0 && threadIdx.x == 0) {
        float m = attn[0];
        for (int i = 1; i < P_; ++i) m = fmaxf(m, attn[i]);
        float e[P_], s = 0.0f;
        for (int i = 0; i < P_; ++i) { e[i] = __expf(attn[i] - m); s += e[i]; }
        for (int i = 0; i < P_; ++i) probs[i] = e[i] / s;
    }
}

// out[i] = hid[i,:] . w + b   (tiny K=256, N=1 head)
__global__ void lin2_kernel(const __bf16* __restrict__ hid,
                            const float* __restrict__ w,
                            const float* __restrict__ b,
                            float* __restrict__ out) {
    int i = blockIdx.x * blockDim.x + threadIdx.x;
    if (i >= N_) return;
    float s = b[0];
    const __bf16* hp = hid + (long)i * HID_;
    for (int j = 0; j < HID_; ++j) s += (float)hp[j] * w[j];
    out[i] = s;
}

// ---------------------------------------------------------------------------
// Host-side helpers
// ---------------------------------------------------------------------------
static inline void launch_gemm(hipStream_t st, const __bf16* A, int lda,
                               const __bf16* Bt, const float* bias,
                               float* Cf, int ldcf, __bf16* Cb, int ldcb,
                               int M, int Ncols, int K, int act) {
    dim3 g((M + BM - 1) / BM, Ncols / BN), b(256);
    gemm_bf16_wmma_kernel<<<g, b, 0, st>>>(A, lda, Bt, K, bias, Cf, ldcf,
                                           Cb, ldcb, M, Ncols, K, act);
}

static inline void launch_zero(hipStream_t st, float* p, long n) {
    zero_f32_kernel<<<(int)((n + 255) / 256), 256, 0, st>>>(p, n);
}

struct GcnCtx {
    __bf16* tb; float* accv;
    const int* src; const int* dst;
    const float* normv; const float* selfn;
};

static void run_gcn(hipStream_t st, const GcnCtx& cx,
                    const __bf16* Ain, int lda, int K,
                    const __bf16* Wt, const float* bias,
                    __bf16* outB, int ldOutB, float* outF) {
    launch_gemm(st, Ain, lda, Wt, nullptr, nullptr, 0, cx.tb, C_,
                N_, C_, K, ACT_NONE);
    launch_zero(st, cx.accv, (long)N_ * C_);
    long scat = (long)E_ * (C_ / 4);
    scatter_kernel<<<(int)((scat + 255) / 256), 256, 0, st>>>(
        cx.tb, cx.src, cx.dst, cx.normv, cx.accv);
    long nc = (long)N_ * C_;
    gcn_finish_kernel<<<(int)((nc + 255) / 256), 256, 0, st>>>(
        cx.accv, cx.tb, cx.selfn, bias, outB, ldOutB, outF);
}

static inline void cvt_wt(hipStream_t st, const float* in, __bf16* out,
                          int K, int Nc) {
    long n = (long)K * Nc;
    f32_to_bf16_t_kernel<<<(int)((n + 255) / 256), 256, 0, st>>>(in, out, K, Nc);
}

extern "C" void kernel_launch(void* const* d_in, const int* in_sizes, int n_in,
                              void* d_out, int out_size, void* d_ws, size_t ws_size,
                              hipStream_t stream) {
    (void)in_sizes; (void)n_in; (void)out_size; (void)ws_size;
    const float* x     = (const float*)d_in[0];
    const int*   eidx  = (const int*)d_in[1];
    const float* eattr = (const float*)d_in[2];
    const float* Wf[5]  = {(const float*)d_in[3], (const float*)d_in[5],
                           (const float*)d_in[7], (const float*)d_in[9],
                           (const float*)d_in[11]};
    const float* bf5[5] = {(const float*)d_in[4], (const float*)d_in[6],
                           (const float*)d_in[8], (const float*)d_in[10],
                           (const float*)d_in[12]};
    const float* Wzf = (const float*)d_in[13]; const float* bzf = (const float*)d_in[14];
    const float* Lzw = (const float*)d_in[15]; const float* Lzb = (const float*)d_in[16];
    const float* Wrf = (const float*)d_in[17]; const float* brf = (const float*)d_in[18];
    const float* Lrw = (const float*)d_in[19]; const float* Lrb = (const float*)d_in[20];
    const float* Whf = (const float*)d_in[21]; const float* bhf = (const float*)d_in[22];
    const float* Lhw = (const float*)d_in[23]; const float* Lhb = (const float*)d_in[24];
    const float* attn = (const float*)d_in[25];
    const float* l1w = (const float*)d_in[26]; const float* l1b = (const float*)d_in[27];
    const float* l2w = (const float*)d_in[28]; const float* l2b = (const float*)d_in[29];

    const int* srcI = eidx;        // edge_index[0]
    const int* dstI = eidx + E_;   // edge_index[1]

    float* outHead = (float*)d_out;        // (N,1)
    float* Hacc    = (float*)d_out + N_;   // (N,C) out_hidden

    // ---- carve workspace ----
    char* wp = (char*)d_ws;
    auto take = [&](size_t bytes) -> void* {
        void* p = (void*)wp;
        wp += (bytes + 255) & ~(size_t)255;
        return p;
    };
    __bf16* wb[5];
    wb[0] = (__bf16*)take((size_t)F_ * C_ * 2);
    for (int i = 1; i < 5; ++i) wb[i] = (__bf16*)take((size_t)C_ * C_ * 2);
    __bf16* wbz = (__bf16*)take((size_t)F_ * C_ * 2);
    __bf16* wbr = (__bf16*)take((size_t)F_ * C_ * 2);
    __bf16* wbh = (__bf16*)take((size_t)F_ * C_ * 2);
    __bf16* lbz = (__bf16*)take((size_t)2 * C_ * C_ * 2);
    __bf16* lbr = (__bf16*)take((size_t)2 * C_ * C_ * 2);
    __bf16* lbh = (__bf16*)take((size_t)2 * C_ * C_ * 2);
    __bf16* lb1 = (__bf16*)take((size_t)C_ * HID_ * 2);
    float* deg   = (float*)take((size_t)N_ * 4);
    float* dinv  = (float*)take((size_t)N_ * 4);
    float* normv = (float*)take((size_t)E_ * 4);
    float* selfn = (float*)take((size_t)N_ * 4);
    float* probs = (float*)take(4 * 4);
    __bf16* Xp  = (__bf16*)take((size_t)N_ * F_ * 2);
    __bf16* tb  = (__bf16*)take((size_t)N_ * C_ * 2);
    float*  accv = (float*)take((size_t)N_ * C_ * 4);
    __bf16* hA  = (__bf16*)take((size_t)N_ * C_ * 2);
    float*  Hf  = (float*)take((size_t)N_ * C_ * 4);
    __bf16* CC  = (__bf16*)take((size_t)N_ * 2 * C_ * 2);
    float*  Zf  = (float*)take((size_t)N_ * C_ * 4);
    float*  Rf  = (float*)take((size_t)N_ * C_ * 4);
    float*  Htf = (float*)take((size_t)N_ * C_ * 4);
    __bf16* hid = (__bf16*)take((size_t)N_ * HID_ * 2);

    // ---- one-time prep: transposed bf16 weights (fragment-friendly B) ----
    cvt_wt(stream, Wf[0], wb[0], F_, C_);
    for (int i = 1; i < 5; ++i) cvt_wt(stream, Wf[i], wb[i], C_, C_);
    cvt_wt(stream, Wzf, wbz, F_, C_);
    cvt_wt(stream, Wrf, wbr, F_, C_);
    cvt_wt(stream, Whf, wbh, F_, C_);
    cvt_wt(stream, Lzw, lbz, 2 * C_, C_);
    cvt_wt(stream, Lrw, lbr, 2 * C_, C_);
    cvt_wt(stream, Lhw, lbh, 2 * C_, C_);
    cvt_wt(stream, l1w, lb1, C_, HID_);

    launch_zero(stream, deg, N_);
    deg_scatter_kernel<<<(E_ + 255) / 256, 256, 0, stream>>>(eattr, dstI, deg);
    dinv_kernel<<<(N_ + 255) / 256, 256, 0, stream>>>(deg, dinv, selfn);
    edge_norm_kernel<<<(E_ + 255) / 256, 256, 0, stream>>>(eattr, srcI, dstI,
                                                           dinv, normv);
    softmax_attn_kernel<<<1, 32, 0, stream>>>(attn, probs);
    launch_zero(stream, Hacc, (long)N_ * C_);

    GcnCtx cx{tb, accv, srcI, dstI, normv, selfn};
    const long nc = (long)N_ * C_;
    const int ncBlocks = (int)((nc + 255) / 256);

    for (int p = 0; p < P_; ++p) {
        slice_xp_kernel<<<(int)(((long)N_ * F_ + 255) / 256), 256, 0, stream>>>(x, Xp, p);

        // 5-layer GCN chain -> H (fp32 in Hf, bf16 scratch in hA)
        run_gcn(stream, cx, Xp, F_, F_, wb[0], bf5[0], hA, C_, nullptr);
        run_gcn(stream, cx, hA, C_, C_, wb[1], bf5[1], hA, C_, nullptr);
        run_gcn(stream, cx, hA, C_, C_, wb[2], bf5[2], hA, C_, nullptr);
        run_gcn(stream, cx, hA, C_, C_, wb[3], bf5[3], hA, C_, nullptr);
        run_gcn(stream, cx, hA, C_, C_, wb[4], bf5[4], hA, C_, Hf);

        // Z gate: CC = [gcn(Xp,Wz) | H] ; Z = sigmoid(CC @ Lz + b)
        run_gcn(stream, cx, Xp, F_, F_, wbz, bzf, CC, 2 * C_, nullptr);
        concat_right_kernel<<<ncBlocks, 256, 0, stream>>>(Hf, nullptr, CC);
        launch_gemm(stream, CC, 2 * C_, lbz, Lzb, Zf, C_, nullptr, 0,
                    N_, C_, 2 * C_, ACT_SIG);

        // R gate
        run_gcn(stream, cx, Xp, F_, F_, wbr, brf, CC, 2 * C_, nullptr);
        concat_right_kernel<<<ncBlocks, 256, 0, stream>>>(Hf, nullptr, CC);
        launch_gemm(stream, CC, 2 * C_, lbr, Lrb, Rf, C_, nullptr, 0,
                    N_, C_, 2 * C_, ACT_SIG);

        // Candidate: CC = [gcn(Xp,Wh) | H*R] ; Ht = tanh(CC @ Lh + b)
        run_gcn(stream, cx, Xp, F_, F_, wbh, bhf, CC, 2 * C_, nullptr);
        concat_right_kernel<<<ncBlocks, 256, 0, stream>>>(Hf, Rf, CC);
        launch_gemm(stream, CC, 2 * C_, lbh, Lhb, Htf, C_, nullptr, 0,
                    N_, C_, 2 * C_, ACT_TANH);

        gate_kernel<<<ncBlocks, 256, 0, stream>>>(Zf, Hf, Htf, probs, p, Hacc);
    }

    // MLP head: relu(Hacc) @ lin1 (relu) -> hid ; hid @ lin2 -> outHead
    relu_to_bf16_kernel<<<ncBlocks, 256, 0, stream>>>(Hacc, hA);
    launch_gemm(stream, hA, C_, lb1, l1b, nullptr, 0, hid, HID_,
                N_, HID_, C_, ACT_RELU);
    lin2_kernel<<<(N_ + 255) / 256, 256, 0, stream>>>(hid, l2w, l2b, outHead);
}
